// SVHTwoTensorPermEquivariant_13228499271738
// MI455X (gfx1250) — matrix operand order repaired and checked by reading, not compiled, not run on hardware
//
#include <hip/hip_runtime.h>
#include <hip/hip_bf16.h>

typedef __attribute__((ext_vector_type(16))) _Float16 v16h;
typedef __attribute__((ext_vector_type(8)))  _Float16 v8h;
typedef __attribute__((ext_vector_type(8)))  float    v8f;

__device__ __forceinline__ v16h concat8(v8h lo, v8h hi) {
    return __builtin_shufflevector(lo, hi, 0,1,2,3,4,5,6,7,8,9,10,11,12,13,14,15);
}

__device__ __forceinline__ v8h pack8(v8f r) {
    v8h h;
#pragma unroll
    for (int i = 0; i < 8; ++i) h[i] = (_Float16)r[i];
    return h;
}

// A-operand (16x32 f16) load from row-major [mt][16 rows][32 K] LDS block.
// lane<16: K chunks {0,1} (K=0..15); lane>=16: chunks {2,3} (K=16..31)
__device__ __forceinline__ v16h load_wA(const _Float16* Wbase, int mt, int row, int hf) {
    const v8h* p = (const v8h*)(Wbase + ((mt * 16 + row) << 5));
    return concat8(p[2 * hf], p[2 * hf + 1]);
}

__device__ __forceinline__ void eigvec3(float a00, float a01, float a02,
                                        float a11, float a12, float a22,
                                        float lam, float v[3]) {
    float m00 = a00 - lam, m11 = a11 - lam, m22 = a22 - lam;
    float c0x = a01 * a12 - a02 * m11, c0y = a02 * a01 - m00 * a12, c0z = m00 * m11 - a01 * a01;
    float c1x = a01 * m22 - a02 * a12, c1y = a02 * a02 - m00 * m22, c1z = m00 * a12 - a01 * a02;
    float c2x = m11 * m22 - a12 * a12, c2y = a12 * a02 - a01 * m22, c2z = a01 * a12 - m11 * a02;
    float n0 = c0x * c0x + c0y * c0y + c0z * c0z;
    float n1 = c1x * c1x + c1y * c1y + c1z * c1z;
    float n2 = c2x * c2x + c2y * c2y + c2z * c2z;
    float vx = c0x, vy = c0y, vz = c0z, nm = n0;
    if (n1 > nm) { vx = c1x; vy = c1y; vz = c1z; nm = n1; }
    if (n2 > nm) { vx = c2x; vy = c2y; vz = c2z; nm = n2; }
    if (nm < 1e-30f) { vx = 1.f; vy = 0.f; vz = 0.f; nm = 1.f; }
    float inv = rsqrtf(nm);
    v[0] = vx * inv; v[1] = vy * inv; v[2] = vz * inv;
}

__global__ __launch_bounds__(128)
void perm_equiv_eig_kernel(const float* __restrict__ x,
                           const float* __restrict__ wid0, const float* __restrict__ wsum0, const float* __restrict__ b0,
                           const float* __restrict__ wid1, const float* __restrict__ wsum1, const float* __restrict__ b1,
                           const float* __restrict__ wid2, const float* __restrict__ wsum2, const float* __restrict__ b2,
                           const float* __restrict__ wid3, const float* __restrict__ wsum3, const float* __restrict__ b3,
                           float* __restrict__ out, int n) {
    // Weights for layers 1,2 staged in f16, padded 23->32 both dims.
    // sW[mat][mtile][row 0..15][K 0..31], mat: 0=id1 1=sum1 2=id2 3=sum2
    alignas(64) __shared__ _Float16 sW[4][2][16][32];
    // Activations: [wave][group of 16 elems][slot u: d0,d1,d2,S][elem][ch 0..31]
    alignas(64) __shared__ _Float16 sAct[4][2][4][16][32];

    const int tid = threadIdx.x;

    // ---- cooperative weight prep (once per block) ----
    for (int idx = tid; idx < 4 * 2 * 16 * 32; idx += 128) {
        int m   = idx >> 10;
        int mt  = (idx >> 9) & 1;
        int row = (idx >> 5) & 15;
        int k   = idx & 31;
        int och = mt * 16 + row;
        const float* ws = (m == 0) ? wid1 : (m == 1) ? wsum1 : (m == 2) ? wid2 : wsum2;
        float v = (och < 23 && k < 23) ? ws[och * 23 + k] : 0.0f;
        sW[m][mt][row][k] = (_Float16)v;
    }
    __syncthreads();

    const int wv   = tid >> 5;
    const int lane = tid & 31;
    const int hf   = lane >> 4;   // half-wave id
    const int e    = lane & 15;   // element within group / matrix row
    const int g    = hf;          // this lane's own element group

    int elem = blockIdx.x * 128 + tid;
    int eL = (elem < n) ? elem : (n - 1);

    // ---- load symmetric 3x3 ----
    const float* xp = x + (size_t)eL * 9;
    float a00 = xp[0], a01 = xp[1], a02 = xp[2];
    float a11 = xp[4], a12 = xp[5], a22 = xp[8];

    // ---- closed-form symmetric 3x3 eigendecomposition ----
    float q  = (a00 + a11 + a22) * (1.0f / 3.0f);
    float d0 = a00 - q, d1 = a11 - q, d2 = a22 - q;
    float p2 = d0 * d0 + d1 * d1 + d2 * d2 + 2.0f * (a01 * a01 + a02 * a02 + a12 * a12);
    float p  = sqrtf(p2 * (1.0f / 6.0f));
    float detB = d0 * (d1 * d2 - a12 * a12)
               - a01 * (a01 * d2 - a12 * a02)
               + a02 * (a01 * a12 - d1 * a02);
    float pinv = (p > 1e-20f) ? (1.0f / p) : 0.0f;
    float r = 0.5f * detB * pinv * pinv * pinv;
    r = fminf(1.0f, fmaxf(-1.0f, r));
    float phi = acosf(r) * (1.0f / 3.0f);
    float l0 = q + 2.0f * p * cosf(phi);
    float l2 = q + 2.0f * p * cosf(phi + 2.0943951023931953f);
    float l1 = 3.0f * q - l0 - l2;

    float V0[3], V1[3], V2[3];
    eigvec3(a00, a01, a02, a11, a12, a22, l0, V0);
    eigvec3(a00, a01, a02, a11, a12, a22, l1, V1);
    V2[0] = V0[1] * V1[2] - V0[2] * V1[1];
    V2[1] = V0[2] * V1[0] - V0[0] * V1[2];
    V2[2] = V0[0] * V1[1] - V0[1] * V1[0];
    {
        float nm = V2[0] * V2[0] + V2[1] * V2[1] + V2[2] * V2[2];
        float inv = rsqrtf(fmaxf(nm, 1e-30f));
        V2[0] *= inv; V2[1] *= inv; V2[2] *= inv;
    }

    // ---- layer 0 (1 -> 23) in VALU, store f16 activations in B-operand layout ----
    float ssum = l0 + l1 + l2;
#pragma unroll
    for (int c = 0; c < 32; ++c) {
        float wi = (c < 23) ? wid0[c]  : 0.0f;
        float ws = (c < 23) ? wsum0[c] : 0.0f;
        float bb = (c < 23) ? b0[c]    : 0.0f;
        float y0 = fmaxf(0.0f, wi * l0 + ws * ssum + bb);
        float y1 = fmaxf(0.0f, wi * l1 + ws * ssum + bb);
        float y2 = fmaxf(0.0f, wi * l2 + ws * ssum + bb);
        sAct[wv][g][0][e][c] = (_Float16)y0;
        sAct[wv][g][1][e][c] = (_Float16)y1;
        sAct[wv][g][2][e][c] = (_Float16)y2;
        sAct[wv][g][3][e][c] = (_Float16)(y0 + y1 + y2);
    }
    __syncthreads();

    // ---- layers 1,2 (23 -> 23) on the matrix engine ----
#pragma unroll
    for (int L = 0; L < 2; ++L) {
        const float*    bptr = L ? b2 : b1;
        const _Float16* Wid  = &sW[2 * L + 0][0][0][0];
        const _Float16* Wsum = &sW[2 * L + 1][0][0][0];
#pragma unroll
        for (int gg = 0; gg < 2; ++gg) {
            // B operands: 16 contiguous K-halves per lane, N = element
            v16h T0 = *(const v16h*)&sAct[wv][gg][0][e][hf * 16];
            v16h T1 = *(const v16h*)&sAct[wv][gg][1][e][hf * 16];
            v16h T2 = *(const v16h*)&sAct[wv][gg][2][e][hf * 16];
            v16h Ts = *(const v16h*)&sAct[wv][gg][3][e][hf * 16];
#pragma unroll
            for (int mt = 0; mt < 2; ++mt) {
                // bias into the C accumulator: och = mt*16 + hf*8 + v
                v8f acc;
#pragma unroll
                for (int v = 0; v < 8; ++v) {
                    int och = mt * 16 + hf * 8 + v;
                    acc[v] = (och < 23) ? bptr[och] : 0.0f;
                }
                v16h Asum = load_wA(Wsum, mt, e, hf);
                acc = __builtin_amdgcn_wmma_f32_16x16x32_f16(false, Asum, false, Ts,
                                                             (short)0, acc, false, false);
                v16h Aid = load_wA(Wid, mt, e, hf);
                v8f y0 = __builtin_amdgcn_wmma_f32_16x16x32_f16(false, Aid, false, T0,
                                                                (short)0, acc, false, false);
                v8f y1 = __builtin_amdgcn_wmma_f32_16x16x32_f16(false, Aid, false, T1,
                                                                (short)0, acc, false, false);
                v8f y2 = __builtin_amdgcn_wmma_f32_16x16x32_f16(false, Aid, false, T2,
                                                                (short)0, acc, false, false);
                v8f r0, r1, r2, sn;
#pragma unroll
                for (int v = 0; v < 8; ++v) {
                    r0[v] = fmaxf(y0[v], 0.0f);
                    r1[v] = fmaxf(y1[v], 0.0f);
                    r2[v] = fmaxf(y2[v], 0.0f);
                    sn[v] = r0[v] + r1[v] + r2[v];
                }
                int ck = 2 * mt + hf; // 8-half chunk within the 32-ch row
                ((v8h*)&sAct[wv][gg][0][e][0])[ck] = pack8(r0);
                ((v8h*)&sAct[wv][gg][1][e][0])[ck] = pack8(r1);
                ((v8h*)&sAct[wv][gg][2][e][0])[ck] = pack8(r2);
                ((v8h*)&sAct[wv][gg][3][e][0])[ck] = pack8(sn);
            }
        }
        __syncthreads();
    }

    // ---- layer 3 (23 -> 1) in VALU ----
    float e0 = 0.0f, e1 = 0.0f, e2 = 0.0f, sa = 0.0f;
    const _Float16* t0p = &sAct[wv][g][0][e][0];
    const _Float16* t1p = &sAct[wv][g][1][e][0];
    const _Float16* t2p = &sAct[wv][g][2][e][0];
    const _Float16* tsp = &sAct[wv][g][3][e][0];
#pragma unroll
    for (int c = 0; c < 23; ++c) {
        float wi = wid3[c], ws = wsum3[c];
        e0 += wi * (float)t0p[c];
        e1 += wi * (float)t1p[c];
        e2 += wi * (float)t2p[c];
        sa += ws * (float)tsp[c];
    }
    float bb3 = b3[0];
    e0 += sa + bb3; e1 += sa + bb3; e2 += sa + bb3;

    // ---- reconstruct V diag(ev) V^T ----
    if (elem < n) {
        float* op = out + (size_t)elem * 9;
#pragma unroll
        for (int i = 0; i < 3; ++i)
#pragma unroll
            for (int j = 0; j < 3; ++j)
                op[i * 3 + j] = e0 * V0[i] * V0[j] + e1 * V1[i] * V1[j] + e2 * V2[i] * V2[j];
    }
}

extern "C" void kernel_launch(void* const* d_in, const int* in_sizes, int n_in,
                              void* d_out, int out_size, void* d_ws, size_t ws_size,
                              hipStream_t stream) {
    const float* x     = (const float*)d_in[0];
    const float* wid0  = (const float*)d_in[1];
    const float* wsum0 = (const float*)d_in[2];
    const float* b0    = (const float*)d_in[3];
    const float* wid1  = (const float*)d_in[4];
    const float* wsum1 = (const float*)d_in[5];
    const float* b1    = (const float*)d_in[6];
    const float* wid2  = (const float*)d_in[7];
    const float* wsum2 = (const float*)d_in[8];
    const float* b2    = (const float*)d_in[9];
    const float* wid3  = (const float*)d_in[10];
    const float* wsum3 = (const float*)d_in[11];
    const float* b3    = (const float*)d_in[12];

    int n = in_sizes[0] / 9;
    int blocks = (n + 127) / 128;
    perm_equiv_eig_kernel<<<blocks, 128, 0, stream>>>(
        x, wid0, wsum0, b0, wid1, wsum1, b1, wid2, wsum2, b2,
        wid3, wsum3, b3, (float*)d_out, n);
}
